// FC_mono_12086037971055
// MI455X (gfx1250) — matrix operand order repaired and checked by reading, not compiled, not run on hardware
//
#include <hip/hip_runtime.h>
#include <hip/hip_bf16.h>

// ---------------- constants ----------------
enum { Bb = 32, Tt = 4096, Cc = 256 };
#define BN_EPS 1e-3f

typedef __attribute__((ext_vector_type(16))) __bf16        v16bf;
typedef __attribute__((ext_vector_type(8)))  float         v8f;
typedef __attribute__((ext_vector_type(4)))  unsigned int  u32x4;
typedef __attribute__((ext_vector_type(2)))  float         f32x2;

union FragBF { v16bf v; u32x4 q[2]; };
union AccF   { v8f   v; float f[8]; };

// float -> bf16 bits, round-to-nearest-even
static __device__ __forceinline__ unsigned short f2bf(float f) {
    unsigned int u = __float_as_uint(f);
    u += 0x7FFFu + ((u >> 16) & 1u);
    return (unsigned short)(u >> 16);
}
static __device__ __forceinline__ unsigned int pack2bf(float lo, float hi) {
    return (unsigned int)f2bf(lo) | ((unsigned int)f2bf(hi) << 16);
}

// ---------------- tiny MLP (scalar) ----------------
static __device__ __forceinline__ float mlp_scalar(
    float z,
    const float* __restrict__ W1, const float* __restrict__ b1,
    const float* __restrict__ W2, const float* __restrict__ b2,
    const float* __restrict__ gamma, const float* __restrict__ beta,
    const float* __restrict__ mmean, const float* __restrict__ mvar,
    const float* __restrict__ Wp, const float* __restrict__ bp)
{
    float h1[4], h2[4];
#pragma unroll
    for (int j = 0; j < 4; ++j) {
        float t = z * W1[j] + b1[j];
        h1[j] = (t > 0.f) ? t : (__expf(t) - 1.f);           // elu
    }
#pragma unroll
    for (int j = 0; j < 4; ++j) {
        float t = b2[j];
#pragma unroll
        for (int i = 0; i < 4; ++i) t += h1[i] * W2[i * 4 + j];
        h2[j] = (t > 0.f) ? t : (__expf(t) - 1.f);           // elu
    }
    float acc = bp[0];
#pragma unroll
    for (int j = 0; j < 4; ++j) {
        float s  = rsqrtf(mvar[j] + BN_EPS) * gamma[j];
        float bn = (h2[j] - mmean[j]) * s + beta[j];
        acc += bn * Wp[j];
    }
    return 1.f / (1.f + __expf(-acc));                        // sigmoid
}

// ---------------- 1) column sum of squares over T ----------------
__global__ __launch_bounds__(256) void colsumsq_kernel(
    const float* __restrict__ x, float* __restrict__ sumsq)
{
    const int b = blockIdx.x, ch = blockIdx.y, c = threadIdx.x;
    const float* p = x + (size_t)b * Tt * Cc + (size_t)ch * 256 * Cc + c;
    float acc = 0.f;
    for (int t = 0; t < 256; ++t) { float v = p[(size_t)t * Cc]; acc += v * v; }
    atomicAdd(&sumsq[b * Cc + c], acc);
}

// ---------------- 2) sumsq -> 1/sqrt(sumsq + 1e-7) in place ----------------
__global__ void invnorm_kernel(float* __restrict__ s)
{
    int i = blockIdx.x * 256 + threadIdx.x;
    s[i] = rsqrtf(s[i] + 1e-7f);
}

// ---------------- 3) cor[b,c,d] = sum_t nx[t,c]*nx[t,d]  (bf16 WMMA) ----------------
// block region 64(c) x 64(d); 8 waves, 2 tiles/wave sharing the A fragment.
// K loop over T in 32-chunks, LDS double-buffered (one barrier per step).
__global__ __launch_bounds__(256) void corr_gemm_kernel(
    const float* __restrict__ x, const float* __restrict__ invn,
    float* __restrict__ cor)
{
    __shared__ unsigned short Sc[2][64 * 40];   // [c][t] transposed, stride 40
    __shared__ unsigned short Sd[2][64 * 40];   // [d][t] transposed

    const int b  = blockIdx.z;
    const int c0 = blockIdx.x * 64;
    const int d0 = blockIdx.y * 64;
    const int tid  = threadIdx.x;
    const int lane = tid & 31;
    const int w    = tid >> 5;
    const int ct  = w & 3;          // A subtile (c)
    const int dtp = w >> 2;         // B subtile pair (d)

    const float* xb = x + (size_t)b * Tt * Cc;
    const int   js   = tid & 63;            // staged column (c or d)
    const int   p0   = tid >> 6;            // t-pair group 0..3
    const float invc = invn[b * Cc + c0 + js];
    const float invd = invn[b * Cc + d0 + js];

    AccF acc0, acc1;
#pragma unroll
    for (int r = 0; r < 8; ++r) { acc0.f[r] = 0.f; acc1.f[r] = 0.f; }

    const int mrow = ct * 16 + (lane & 15);
    const int n0   = dtp * 32 + (lane & 15);
    const int n1   = n0 + 16;
    const int kbA  = (lane < 16) ? 0 : 16;
    const int kbB  = (lane < 16) ? 0 : 16;

    // ---- stage one 32-deep K chunk into buffer `buf` ----
    auto stage = [&](int buf, int t0) {
#pragma unroll
        for (int e = 0; e < 4; ++e) {
            int i = 2 * (p0 + e * 4);                 // even t offset 0..30
            const float* pc = &xb[(size_t)(t0 + i) * Cc + c0 + js];
            float v0 = pc[0]  * invc;
            float v1 = pc[Cc] * invc;
            *(unsigned int*)&Sc[buf][js * 40 + i] = pack2bf(v0, v1);
        }
#pragma unroll
        for (int e = 0; e < 4; ++e) {
            int i = 2 * (p0 + e * 4);
            const float* pd = &xb[(size_t)(t0 + i) * Cc + d0 + js];
            float v0 = pd[0]  * invd;
            float v1 = pd[Cc] * invd;
            *(unsigned int*)&Sd[buf][js * 40 + i] = pack2bf(v0, v1);
        }
    };

    stage(0, 0);
    const int NSTEP = Tt / 32;                        // 128
    for (int k = 0; k < NSTEP; ++k) {
        __syncthreads();
        const int buf = k & 1;
        if (k + 1 < NSTEP) {
            stage(buf ^ 1, (k + 1) * 32);
            if (k + 2 < NSTEP)                        // prefetch chunk after next
                __builtin_prefetch(&xb[(size_t)(k + 2) * 32 * Cc + c0 + js], 0, 1);
        }
        FragBF A, B0, B1;
        A.q[0]  = *(const u32x4*)&Sc[buf][mrow * 40 + kbA];
        A.q[1]  = *(const u32x4*)&Sc[buf][mrow * 40 + kbA + 8];
        B0.q[0] = *(const u32x4*)&Sd[buf][n0 * 40 + kbB];
        B0.q[1] = *(const u32x4*)&Sd[buf][n0 * 40 + kbB + 8];
        B1.q[0] = *(const u32x4*)&Sd[buf][n1 * 40 + kbB];
        B1.q[1] = *(const u32x4*)&Sd[buf][n1 * 40 + kbB + 8];
        acc0.v = __builtin_amdgcn_wmma_f32_16x16x32_bf16(
            false, A.v, false, B0.v, (short)0, acc0.v, false, false);
        acc1.v = __builtin_amdgcn_wmma_f32_16x16x32_bf16(
            false, A.v, false, B1.v, (short)0, acc1.v, false, false);
    }

    const int nn    = lane & 15;
    const int mbase = (lane >> 4) * 8;
    float* op0 = cor + ((size_t)b * Cc + c0 + ct * 16) * Cc + d0 + dtp * 32;
#pragma unroll
    for (int r = 0; r < 8; ++r) {
        op0[(size_t)(mbase + r) * Cc + nn]      = acc0.f[r];
        op0[(size_t)(mbase + r) * Cc + nn + 16] = acc1.f[r];
    }
}

// ---------------- 4) att = mlp(cor), in place ----------------
__global__ __launch_bounds__(256) void att_kernel(
    float* __restrict__ cor,
    const float* W1, const float* b1, const float* W2, const float* b2,
    const float* gamma, const float* beta, const float* mmean,
    const float* mvar, const float* Wp, const float* bp, int n)
{
    int i = blockIdx.x * 256 + threadIdx.x;
    if (i < n)
        cor[i] = mlp_scalar(cor[i], W1, b1, W2, b2, gamma, beta, mmean, mvar, Wp, bp);
}

// ---------------- 5) invsum[b,d] = 1/(sum_c att[b,c,d] + 1e-7) ----------------
__global__ __launch_bounds__(256) void colsum_kernel(
    const float* __restrict__ att, float* __restrict__ invsum)
{
    const int b = blockIdx.x, d = threadIdx.x;
    const float* p = att + (size_t)b * Cc * Cc + d;
    float s = 0.f;
    for (int c = 0; c < Cc; ++c) s += p[(size_t)c * Cc];
    invsum[b * Cc + d] = 1.f / (s + 1e-7f);
}

// ---------------- 6) Mbf[b,c,d] = bf16( (c==d?0:att) * invsum[b,d] ) ----------------
__global__ __launch_bounds__(256) void maskcvt_kernel(
    const float* __restrict__ att, const float* __restrict__ invsum,
    unsigned short* __restrict__ Mbf)
{
    int idx = blockIdx.x * 256 + threadIdx.x;     // over B*C*C = 2^21
    int b   = idx >> 16;                          // C*C = 65536
    int rem = idx & 65535;
    int c = rem >> 8, d = rem & 255;
    float v = (c == d) ? 0.f : att[idx] * invsum[b * Cc + d];
    Mbf[idx] = f2bf(v);
}

// ---------------- 7) out = x + x @ M  (bf16 WMMA) ----------------
// block region 64(t) x 64(d); 8 waves, 2 tiles/wave sharing the A fragment.
// K loop over C in 32-chunks, LDS double-buffered.
__global__ __launch_bounds__(256) void apply_gemm_kernel(
    const float* __restrict__ x, const unsigned short* __restrict__ Mbf,
    float* __restrict__ out)
{
    __shared__ unsigned short Xs[2][64 * 40];  // [t][k] row-major, stride 40
    __shared__ unsigned short Ms[2][64 * 40];  // [d][k] transposed

    const int b  = blockIdx.z;
    const int t0 = blockIdx.x * 64;
    const int d0 = blockIdx.y * 64;
    const int tid  = threadIdx.x;
    const int lane = tid & 31;
    const int w    = tid >> 5;
    const int tt  = w & 3;          // A subtile (t)
    const int dtp = w >> 2;         // B subtile pair (d)

    const float* xb = x + (size_t)b * Tt * Cc;
    const unsigned short* Mb = Mbf + (size_t)b * Cc * Cc;

    const int jpx = tid & 15,  ix0 = tid >> 4;   // Xs: k-pair, t row group
    const int jpm = tid & 31,  im0 = tid >> 5;   // Ms: d-pair, k row group

    AccF acc0, acc1;
#pragma unroll
    for (int r = 0; r < 8; ++r) { acc0.f[r] = 0.f; acc1.f[r] = 0.f; }

    const int m    = tt * 16 + (lane & 15);
    const int n0   = dtp * 32 + (lane & 15);
    const int n1   = n0 + 16;
    const int kbA  = (lane < 16) ? 0 : 16;
    const int kbB  = (lane < 16) ? 0 : 16;

    auto stage = [&](int buf, int k0) {
        // Xs: 64 t x 32 k, float2 global loads, packed b32 LDS stores
#pragma unroll
        for (int e = 0; e < 4; ++e) {
            int i = ix0 + e * 16;                                   // t row 0..63
            f32x2 v = *(const f32x2*)&xb[(size_t)(t0 + i) * Cc + k0 + 2 * jpx];
            *(unsigned int*)&Xs[buf][i * 40 + 2 * jpx] = pack2bf(v.x, v.y);
        }
        // Ms: transpose 32 k x 64 d -> [d][k]; b32 loads (2 d), two b16 stores
#pragma unroll
        for (int e = 0; e < 4; ++e) {
            int i = im0 + e * 8;                                    // k row 0..31
            unsigned int u = *(const unsigned int*)&Mb[(size_t)(k0 + i) * Cc + d0 + 2 * jpm];
            Ms[buf][(2 * jpm) * 40 + i]     = (unsigned short)(u & 0xFFFFu);
            Ms[buf][(2 * jpm + 1) * 40 + i] = (unsigned short)(u >> 16);
        }
    };

    stage(0, 0);
    const int NSTEP = Cc / 32;                     // 8
    for (int k = 0; k < NSTEP; ++k) {
        __syncthreads();
        const int buf = k & 1;
        if (k + 1 < NSTEP) stage(buf ^ 1, (k + 1) * 32);
        FragBF A, B0, B1;
        A.q[0]  = *(const u32x4*)&Xs[buf][m * 40 + kbA];
        A.q[1]  = *(const u32x4*)&Xs[buf][m * 40 + kbA + 8];
        B0.q[0] = *(const u32x4*)&Ms[buf][n0 * 40 + kbB];
        B0.q[1] = *(const u32x4*)&Ms[buf][n0 * 40 + kbB + 8];
        B1.q[0] = *(const u32x4*)&Ms[buf][n1 * 40 + kbB];
        B1.q[1] = *(const u32x4*)&Ms[buf][n1 * 40 + kbB + 8];
        acc0.v = __builtin_amdgcn_wmma_f32_16x16x32_bf16(
            false, A.v, false, B0.v, (short)0, acc0.v, false, false);
        acc1.v = __builtin_amdgcn_wmma_f32_16x16x32_bf16(
            false, A.v, false, B1.v, (short)0, acc1.v, false, false);
    }

    const int nn = lane & 15;
    const int mb = (lane >> 4) * 8;
    size_t base = (size_t)b * Tt * Cc + (size_t)(t0 + tt * 16) * Cc + d0 + dtp * 32;
#pragma unroll
    for (int r = 0; r < 8; ++r) {
        size_t i0 = base + (size_t)(mb + r) * Cc + nn;
        // out is write-once and never re-read: bypass caches (non-temporal)
        __builtin_nontemporal_store(x[i0]      + acc0.f[r], &out[i0]);
        __builtin_nontemporal_store(x[i0 + 16] + acc1.f[r], &out[i0 + 16]);
    }
}

// ---------------- 8) monotonicity penalty (single wave) ----------------
__global__ void penalty_kernel(
    const float* W1, const float* b1, const float* W2, const float* b2,
    const float* gamma, const float* beta, const float* mmean,
    const float* mvar, const float* Wp, const float* bp,
    float* __restrict__ outp)
{
    __shared__ float ko[21];
    int i = threadIdx.x;
    if (i < 21) {
        float z = -1.f + 0.1f * (float)i;   // linspace(-1,1,21)
        ko[i] = mlp_scalar(z, W1, b1, W2, b2, gamma, beta, mmean, mvar, Wp, bp);
    }
    __syncthreads();
    if (i == 0) {
        float s = 0.f;
        for (int j = 0; j < 10; ++j) {
            float dL = ko[j + 1]  - ko[j];
            float dR = ko[j + 11] - ko[j + 10];
            s += (fabsf(dL) - dL) + (fabsf(dR) - dR);
        }
        *outp = 0.5f * 10.0f * (s * 0.1f);  // 0.5 * PENALTY_RATE * mean
    }
}

// ---------------- launcher ----------------
extern "C" void kernel_launch(void* const* d_in, const int* in_sizes, int n_in,
                              void* d_out, int out_size, void* d_ws, size_t ws_size,
                              hipStream_t stream)
{
    const float* x     = (const float*)d_in[0];
    const float* W1    = (const float*)d_in[1];
    const float* b1    = (const float*)d_in[2];
    const float* W2    = (const float*)d_in[3];
    const float* b2    = (const float*)d_in[4];
    const float* gamma = (const float*)d_in[5];
    const float* beta  = (const float*)d_in[6];
    const float* mmean = (const float*)d_in[7];
    const float* mvar  = (const float*)d_in[8];
    const float* Wp    = (const float*)d_in[9];
    const float* bp    = (const float*)d_in[10];
    float* out = (float*)d_out;

    char* ws = (char*)d_ws;
    float*          sumsq  = (float*)ws;                               // 32 KiB
    float*          invsum = (float*)(ws + 32768);                     // 32 KiB
    float*          cor    = (float*)(ws + 65536);                     // 8 MiB (cor -> att in place)
    unsigned short* Mbf    = (unsigned short*)(ws + 65536 + 8388608);  // 4 MiB

    (void)in_sizes; (void)n_in; (void)out_size; (void)ws_size;

    hipMemsetAsync(sumsq, 0, (size_t)Bb * Cc * sizeof(float), stream);
    colsumsq_kernel<<<dim3(Bb, Tt / 256), 256, 0, stream>>>(x, sumsq);
    invnorm_kernel<<<(Bb * Cc) / 256, 256, 0, stream>>>(sumsq);
    corr_gemm_kernel<<<dim3(Cc / 64, Cc / 64, Bb), 256, 0, stream>>>(x, sumsq, cor);
    att_kernel<<<(Bb * Cc * Cc) / 256, 256, 0, stream>>>(
        cor, W1, b1, W2, b2, gamma, beta, mmean, mvar, Wp, bp, Bb * Cc * Cc);
    colsum_kernel<<<Bb, 256, 0, stream>>>(cor, invsum);
    maskcvt_kernel<<<(Bb * Cc * Cc) / 256, 256, 0, stream>>>(cor, invsum, Mbf);
    apply_gemm_kernel<<<dim3(Tt / 64, Cc / 64, Bb), 256, 0, stream>>>(x, Mbf, out);
    penalty_kernel<<<1, 32, 0, stream>>>(
        W1, b1, W2, b2, gamma, beta, mmean, mvar, Wp, bp,
        out + (size_t)Bb * Tt * Cc);
}
